// DMSTGCN_53455162966228
// MI455X (gfx1250) — matrix operand order repaired and checked by reading, not compiled, not run on hardware
//
#include <hip/hip_runtime.h>
#include <math.h>

// ---------------- problem constants ----------------
#define B_    64
#define T_    16
#define N_    500
#define NP    512      // padded node count for WMMA (zero pad)
#define S_    2
#define DIMS  40
#define RC    16
#define DC    16
#define SC    8
#define EC    16
#define OUTC  3
#define CIN   48
#define SKIPCH 304
#define BN_SCALE 0.9999950000374997f

typedef __attribute__((ext_vector_type(16))) __bf16 v16bf;
typedef __attribute__((ext_vector_type(8)))  __bf16 v8bf;
typedef __attribute__((ext_vector_type(8)))  float  v8f;

// =====================================================================
// Adjacency construction:  A[b,w,v] = softmax_v(relu(P2 M_b P3^T))
// =====================================================================

// Mb[b,j,k] = sum_i te[b,i] * pk[i,j,k],  te = p_emb[ind[b]]
__global__ __launch_bounds__(256)
void k_adj_M(const float* __restrict__ p_emb, const int* __restrict__ ind,
             const float* __restrict__ pk, float* __restrict__ Mb) {
  const int b = blockIdx.x;
  const float* te = p_emb + (size_t)ind[b] * DIMS;
  for (int e = threadIdx.x; e < DIMS * DIMS; e += 256) {
    const int j = e / DIMS, k = e % DIMS;
    float s = 0.f;
    for (int i = 0; i < DIMS; ++i)
      s += te[i] * pk[((size_t)i * DIMS + j) * DIMS + k];
    Mb[(size_t)b * DIMS * DIMS + e] = s;
  }
}

// tmp[b,w,k] = sum_j p2[w,j] * Mb[b,j,k]
__global__ __launch_bounds__(256)
void k_adj_tmp(const float* __restrict__ p2, const float* __restrict__ Mb,
               float* __restrict__ tmp) {
  int idx = blockIdx.x * 256 + threadIdx.x;
  if (idx >= B_ * N_ * DIMS) return;
  const int k = idx % DIMS;
  const int w = (idx / DIMS) % N_;
  const int b = idx / (DIMS * N_);
  const float* Mbb = Mb + (size_t)b * DIMS * DIMS;
  float s = 0.f;
  for (int j = 0; j < DIMS; ++j)
    s += p2[(size_t)w * DIMS + j] * Mbb[j * DIMS + k];
  tmp[idx] = s;
}

// per-(b,w) row: logits[v] = relu(dot(tmp[b,w,:], p3[v,:])); softmax over v;
// write bf16 row of length NP with zero padding; pad rows (w>=N_) all zero.
__global__ __launch_bounds__(256)
void k_adj_soft(const float* __restrict__ tmp, const float* __restrict__ p3,
                __bf16* __restrict__ A) {
  __shared__ float sl[NP];
  __shared__ float red[256];
  const int tid = threadIdx.x;
  const int b = blockIdx.x / NP;
  const int w = blockIdx.x % NP;
  __bf16* Arow = A + ((size_t)b * NP + w) * NP;
  if (w >= N_) {
    for (int v = tid; v < NP; v += 256) Arow[v] = (__bf16)0.0f;
    return;
  }
  const float* trow = tmp + ((size_t)b * N_ + w) * DIMS;
  float lmax = -1e30f;
  for (int v = tid; v < N_; v += 256) {
    float s = 0.f;
#pragma unroll
    for (int k = 0; k < DIMS; ++k) s += trow[k] * p3[(size_t)v * DIMS + k];
    s = fmaxf(s, 0.0f);
    sl[v] = s;
    lmax = fmaxf(lmax, s);
  }
  red[tid] = lmax; __syncthreads();
  for (int off = 128; off > 0; off >>= 1) {
    if (tid < off) red[tid] = fmaxf(red[tid], red[tid + off]);
    __syncthreads();
  }
  const float mx = red[0];
  __syncthreads();
  float lsum = 0.f;
  for (int v = tid; v < N_; v += 256) {
    float e = expf(sl[v] - mx);
    sl[v] = e; lsum += e;
  }
  red[tid] = lsum; __syncthreads();
  for (int off = 128; off > 0; off >>= 1) {
    if (tid < off) red[tid] += red[tid + off];
    __syncthreads();
  }
  const float inv = 1.0f / red[0];
  for (int v = tid; v < N_; v += 256) Arow[v] = (__bf16)(sl[v] * inv);
  for (int v = N_ + tid; v < NP; v += 256) Arow[v] = (__bf16)0.0f;
}

// =====================================================================
// Start conv: x_s[b,c,n,t] = in[b,t,n,s]*sw[s,c] + sb[s,c]
// =====================================================================
__global__ __launch_bounds__(256)
void k_start(const float* __restrict__ in, const float* __restrict__ sw,
             const float* __restrict__ sb, float* __restrict__ x0,
             float* __restrict__ x1) {
  int idx = blockIdx.x * 256 + threadIdx.x;
  if (idx >= B_ * RC * N_ * T_) return;
  const int t = idx % T_;
  const int n = (idx / T_) % N_;
  const int c = (idx / (T_ * N_)) % RC;
  const int b = idx / (T_ * N_ * RC);
  const float v0 = in[(((size_t)b * T_ + t) * N_ + n) * S_ + 0];
  const float v1 = in[(((size_t)b * T_ + t) * N_ + n) * S_ + 1];
  x0[idx] = v0 * sw[0 * RC + c] + sb[0 * RC + c];
  x1[idx] = v1 * sw[1 * RC + c] + sb[1 * RC + c];
}

// =====================================================================
// Gated dilated causal conv (kernel (1,2)):
//   gx = tanh(dconv_f(x)) * sigmoid(dconv_g(x))
// emits f32 [B,DC,N,Tout] and bf16 node-major [B,DC,Tout,NP] (zero pad)
// =====================================================================
__global__ __launch_bounds__(256)
void k_gated(const float* __restrict__ x, int Tin, int d,
             const float* __restrict__ fw, const float* __restrict__ fb,
             const float* __restrict__ gw, const float* __restrict__ gb,
             float* __restrict__ gx, __bf16* __restrict__ gxT, int Tout) {
  int idx = blockIdx.x * 256 + threadIdx.x;
  if (idx >= B_ * DC * NP * Tout) return;
  const int t = idx % Tout;
  const int n = (idx / Tout) % NP;
  const int o = (idx / (Tout * NP)) % DC;
  const int b = idx / (Tout * NP * DC);
  __bf16* outT = gxT + (((size_t)b * DC + o) * Tout + t) * NP + n;
  if (n >= N_) { *outT = (__bf16)0.0f; return; }
  float f = fb[o], g = gb[o];
#pragma unroll
  for (int c = 0; c < RC; ++c) {
    const float x0v = x[(((size_t)b * RC + c) * N_ + n) * Tin + t];
    const float x1v = x[(((size_t)b * RC + c) * N_ + n) * Tin + t + d];
    f += fw[(o * RC + c) * 2 + 0] * x0v + fw[(o * RC + c) * 2 + 1] * x1v;
    g += gw[(o * RC + c) * 2 + 0] * x0v + gw[(o * RC + c) * 2 + 1] * x1v;
  }
  const float val = tanhf(f) * (1.0f / (1.0f + expf(-g)));
  gx[(((size_t)b * DC + o) * N_ + n) * Tout + t] = val;
  *outT = (__bf16)val;
}

// =====================================================================
// Skip projection into reversed-concat buffer [B,304,N]
// =====================================================================
__global__ __launch_bounds__(256)
void k_skip(const float* __restrict__ gx0, int Tout,
            const float* __restrict__ sw, const float* __restrict__ sb,
            float* __restrict__ skip, int choff) {
  int idx = blockIdx.x * 256 + threadIdx.x;
  if (idx >= B_ * SC * N_ * Tout) return;
  const int t = idx % Tout;
  const int n = (idx / Tout) % N_;
  const int sc = (idx / (Tout * N_)) % SC;
  const int b = idx / (Tout * N_ * SC);
  float v = sb[sc];
#pragma unroll
  for (int c = 0; c < DC; ++c)
    v += sw[sc * DC + c] * gx0[(((size_t)b * DC + c) * N_ + n) * Tout + t];
  skip[((size_t)b * SKIPCH + choff + sc * Tout + t) * N_ + n] = v;
}

// =====================================================================
// Graph diffusion, bf16 WMMA:  Y[b,c,t,w] = sum_v A[b,w,v] * X[b,c,t,v]
// Register-blocked: each wave computes 4 w-tiles (one 16-col m-tile),
// sharing one B-fragment across 4 WMMAs (1.25 KB/WMMA of fragment traffic).
// A block = (b, m-tile); its 8 waves cover all 32 w-tiles and issue
// identical B-fragment loads -> WGP$ hits. EXEC is all-ones everywhere.
// =====================================================================
__device__ __forceinline__ v16bf ldfrag(const __bf16* p) {
  const v8bf lo = *(const v8bf*)(p);
  const v8bf hi = *(const v8bf*)(p + 16);
  v16bf r;
#pragma unroll
  for (int i = 0; i < 8; ++i) { r[i] = lo[i]; r[i + 8] = hi[i]; }
  return r;
}

__global__ __launch_bounds__(256)
void k_diff_wmma(const __bf16* __restrict__ A, const __bf16* __restrict__ X,
                 __bf16* __restrict__ Y, int Tt) {
  const int lane = threadIdx.x & 31;
  const int g    = threadIdx.x >> 5;   // w-group: 4 consecutive w-tiles
  const int mt   = blockIdx.x % Tt;    // m-tile (m = c*Tt + t)
  const int b    = blockIdx.x / Tt;

  // B-fragment column m = (c,t); lane<16 holds K {0..7,16..23}, lane>=16 K {8..15,24..31}
  const int m = mt * 16 + (lane & 15);
  const int c = m / Tt;
  const int t = m - c * Tt;
  const __bf16* Xrow = X + (((size_t)b * DC + c) * Tt + t) * NP;
  const int koffB = (lane < 16) ? 0 : 8;

  // A-fragment rows: row = g*64 + j*16 + lane%16 for w-tile j=0..3
  // lane<16 holds K {0..7,16..23}, lane>=16 holds K {8..15,24..31}
  const __bf16* Abase = A + ((size_t)b * NP + g * 64 + (lane & 15)) * NP;
  const int ksel = (lane < 16) ? 0 : 8;

  v8f acc0 = {}, acc1 = {}, acc2 = {}, acc3 = {};
  for (int k0 = 0; k0 < NP; k0 += 32) {
    __builtin_prefetch(Xrow + k0 + 128, 0, 1);
    __builtin_prefetch(Abase + k0 + 128, 0, 1);
    // one shared B fragment
    const v8bf blo = *(const v8bf*)(Xrow + k0 + koffB);
    const v8bf bhi = *(const v8bf*)(Xrow + k0 + koffB + 16);
    v16bf bfm;
#pragma unroll
    for (int i = 0; i < 8; ++i) { bfm[i] = blo[i]; bfm[i + 8] = bhi[i]; }
    // four A fragments (16*NP-element stride folds into the imm offset)
    const v16bf a0 = ldfrag(Abase + (size_t)0 * 16 * NP + k0 + ksel);
    const v16bf a1 = ldfrag(Abase + (size_t)1 * 16 * NP + k0 + ksel);
    const v16bf a2 = ldfrag(Abase + (size_t)2 * 16 * NP + k0 + ksel);
    const v16bf a3 = ldfrag(Abase + (size_t)3 * 16 * NP + k0 + ksel);
    acc0 = __builtin_amdgcn_wmma_f32_16x16x32_bf16(false, a0, false, bfm,
                                                   (short)0, acc0, false, false);
    acc1 = __builtin_amdgcn_wmma_f32_16x16x32_bf16(false, a1, false, bfm,
                                                   (short)0, acc1, false, false);
    acc2 = __builtin_amdgcn_wmma_f32_16x16x32_bf16(false, a2, false, bfm,
                                                   (short)0, acc2, false, false);
    acc3 = __builtin_amdgcn_wmma_f32_16x16x32_bf16(false, a3, false, bfm,
                                                   (short)0, acc3, false, false);
  }
  // D layout: lane col = m (fixed), rows = tile_base + (lane<16?0:8) + j
  __bf16* Yp = Y + (((size_t)b * DC + c) * Tt + t) * NP + g * 64 +
               ((lane < 16) ? 0 : 8);
  v8bf o0, o1, o2, o3;
#pragma unroll
  for (int j = 0; j < 8; ++j) {
    o0[j] = (__bf16)acc0[j];
    o1[j] = (__bf16)acc1[j];
    o2[j] = (__bf16)acc2[j];
    o3[j] = (__bf16)acc3[j];
  }
  *(v8bf*)(Yp + 0)  = o0;
  *(v8bf*)(Yp + 16) = o1;
  *(v8bf*)(Yp + 32) = o2;
  *(v8bf*)(Yp + 48) = o3;
}

// =====================================================================
// GCN pointwise over concat [x, A x, A^2 x] (48 -> 16 channels)
// inputs bf16 node-major [B,16,Tt,NP]; outputs f32 [B,16,N,Tt] and/or
// bf16 node-major [B,16,Tt,NP] (zero pad)
// =====================================================================
__global__ __launch_bounds__(256)
void k_gcn_pw(const __bf16* __restrict__ X, const __bf16* __restrict__ H1,
              const __bf16* __restrict__ H2, int Tt,
              const float* __restrict__ W, const float* __restrict__ bias,
              float* __restrict__ outF, __bf16* __restrict__ outT) {
  int idx = blockIdx.x * 256 + threadIdx.x;
  if (idx >= B_ * RC * NP * Tt) return;
  const int t = idx % Tt;
  const int n = (idx / Tt) % NP;
  const int o = (idx / (Tt * NP)) % RC;
  const int b = idx / (Tt * NP * RC);
  if (n >= N_) {
    if (outT) outT[(((size_t)b * RC + o) * Tt + t) * NP + n] = (__bf16)0.0f;
    return;
  }
  float acc = bias[o];
#pragma unroll
  for (int c = 0; c < RC; ++c) {
    const size_t off = (((size_t)b * RC + c) * Tt + t) * NP + n;
    acc += W[o * CIN + c] * (float)X[off];
    acc += W[o * CIN + 16 + c] * (float)H1[off];
    acc += W[o * CIN + 32 + c] * (float)H2[off];
  }
  if (outF) outF[(((size_t)b * RC + o) * N_ + n) * Tt + t] = acc;
  if (outT) outT[(((size_t)b * RC + o) * Tt + t) * NP + n] = (__bf16)acc;
}

// =====================================================================
// Residual + BN update (keeps the source's x1-overwrite bug):
//  x0' = (2*g0 + xap + r0[..,-Tout:]) * BN;  x1' = r1[..,-Tout:] * BN
// =====================================================================
__global__ __launch_bounds__(256)
void k_update(const float* __restrict__ g0, const float* __restrict__ xap,
              const float* __restrict__ x0o, const float* __restrict__ x1o,
              int Tin, int Tout, float* __restrict__ x0n,
              float* __restrict__ x1n) {
  int idx = blockIdx.x * 256 + threadIdx.x;
  if (idx >= B_ * RC * N_ * Tout) return;
  const int t = idx % Tout;
  const int n = (idx / Tout) % N_;
  const int c = (idx / (Tout * N_)) % RC;
  const int b = idx / (Tout * N_ * RC);
  const size_t oldOff = (((size_t)b * RC + c) * N_ + n) * Tin + t + (Tin - Tout);
  x0n[idx] = (2.0f * g0[idx] + xap[idx] + x0o[oldOff]) * BN_SCALE;
  x1n[idx] = x1o[oldOff] * BN_SCALE;
}

// =====================================================================
// End MLP: out = W2 * relu(W1 * relu(skip) + b1) + b2, per (b,n)
// =====================================================================
__global__ __launch_bounds__(256)
void k_end(const float* __restrict__ skip, const float* __restrict__ w1,
           const float* __restrict__ b1, const float* __restrict__ w2,
           const float* __restrict__ b2, float* __restrict__ out) {
  int idx = blockIdx.x * 256 + threadIdx.x;
  if (idx >= B_ * N_) return;
  const int n = idx % N_;
  const int b = idx / N_;
  float acc[EC];
#pragma unroll
  for (int e = 0; e < EC; ++e) acc[e] = b1[e];
  for (int c = 0; c < SKIPCH; ++c) {
    const float sv = fmaxf(skip[((size_t)b * SKIPCH + c) * N_ + n], 0.0f);
#pragma unroll
    for (int e = 0; e < EC; ++e) acc[e] += w1[e * SKIPCH + c] * sv;
  }
#pragma unroll
  for (int e = 0; e < EC; ++e) acc[e] = fmaxf(acc[e], 0.0f);
#pragma unroll
  for (int o = 0; o < OUTC; ++o) {
    float v = b2[o];
#pragma unroll
    for (int e = 0; e < EC; ++e) v += w2[o * EC + e] * acc[e];
    out[((size_t)b * OUTC + o) * N_ + n] = v;
  }
}

// =====================================================================
// Host orchestration
// =====================================================================
extern "C" void kernel_launch(void* const* d_in, const int* in_sizes, int n_in,
                              void* d_out, int out_size, void* d_ws,
                              size_t ws_size, hipStream_t stream) {
  (void)in_sizes; (void)n_in; (void)out_size; (void)ws_size;
  const float* inputs  = (const float*)d_in[0];
  const int*   ind     = (const int*)d_in[1];
  const float* start_w = (const float*)d_in[2];
  const float* start_b = (const float*)d_in[3];
  const float* p1      = (const float*)d_in[4];
  const float* p2      = (const float*)d_in[5];
  const float* p3      = (const float*)d_in[6];
  const float* pk      = (const float*)d_in[7];
  const float* a2p1    = (const float*)d_in[8];
  const float* a2p2    = (const float*)d_in[9];
  const float* a2p3    = (const float*)d_in[10];
  const float* a2pk    = (const float*)d_in[11];
  const float* filt_w  = (const float*)d_in[12];
  const float* filt_b  = (const float*)d_in[13];
  const float* gate_w  = (const float*)d_in[14];
  const float* gate_b  = (const float*)d_in[15];
  const float* gconv_w = (const float*)d_in[16];
  const float* gconv_b = (const float*)d_in[17];
  const float* ga_w    = (const float*)d_in[18];
  const float* ga_b    = (const float*)d_in[19];
  const float* skip_w  = (const float*)d_in[20];
  const float* skip_b  = (const float*)d_in[21];
  const float* end1_w  = (const float*)d_in[22];
  const float* end1_b  = (const float*)d_in[23];
  const float* end2_w  = (const float*)d_in[24];
  const float* end2_b  = (const float*)d_in[25];
  float* out = (float*)d_out;

  // ---- workspace carving (256B aligned) ----
  char* wp = (char*)d_ws;
  auto carve = [&](size_t bytes) -> void* {
    void* r = (void*)wp;
    wp += (bytes + 255) & ~(size_t)255;
    return r;
  };
  const size_t xBytes  = (size_t)B_ * RC * N_ * T_ * sizeof(float);
  const size_t tBytes  = (size_t)B_ * RC * T_ * NP * sizeof(__bf16);
  __bf16* A1    = (__bf16*)carve((size_t)B_ * NP * NP * sizeof(__bf16));
  __bf16* A2    = (__bf16*)carve((size_t)B_ * NP * NP * sizeof(__bf16));
  float*  Mb    = (float*)carve((size_t)B_ * DIMS * DIMS * sizeof(float));
  float*  tmp   = (float*)carve((size_t)B_ * N_ * DIMS * sizeof(float));
  float*  x0a   = (float*)carve(xBytes);
  float*  x0b   = (float*)carve(xBytes);
  float*  x1a   = (float*)carve(xBytes);
  float*  x1b   = (float*)carve(xBytes);
  float*  gx0   = (float*)carve(xBytes);
  float*  gx1   = (float*)carve(xBytes);
  float*  g0f   = (float*)carve(xBytes);
  float*  xapf  = (float*)carve(xBytes);
  __bf16* gx0T  = (__bf16*)carve(tBytes);
  __bf16* gx1T  = (__bf16*)carve(tBytes);
  __bf16* h1T   = (__bf16*)carve(tBytes);
  __bf16* h2T   = (__bf16*)carve(tBytes);
  __bf16* g1T   = (__bf16*)carve(tBytes);
  float*  skipB = (float*)carve((size_t)B_ * SKIPCH * N_ * sizeof(float));

  // ---- adjacency 1 ----
  k_adj_M<<<B_, 256, 0, stream>>>(p1, ind, pk, Mb);
  k_adj_tmp<<<(B_ * N_ * DIMS + 255) / 256, 256, 0, stream>>>(p2, Mb, tmp);
  k_adj_soft<<<B_ * NP, 256, 0, stream>>>(tmp, p3, A1);
  // ---- adjacency 2 ----
  k_adj_M<<<B_, 256, 0, stream>>>(a2p1, ind, a2pk, Mb);
  k_adj_tmp<<<(B_ * N_ * DIMS + 255) / 256, 256, 0, stream>>>(a2p2, Mb, tmp);
  k_adj_soft<<<B_ * NP, 256, 0, stream>>>(tmp, a2p3, A2);

  // ---- start convs ----
  k_start<<<(B_ * RC * N_ * T_ + 255) / 256, 256, 0, stream>>>(
      inputs, start_w, start_b, x0a, x1a);

  float *x0c = x0a, *x1c = x1a, *x0n = x0b, *x1n = x1b;
  int Tin = T_;
  const int dil[4] = {1, 2, 4, 8};
  const int skoff[4] = {184, 80, 8, 0};  // reversed concat offsets

  for (int l = 0; l < 4; ++l) {
    const int d = dil[l];
    const int Tout = Tin - d;
    const int nGated = B_ * DC * NP * Tout;
    const int nPW    = B_ * RC * NP * Tout;
    const int nDiffB = B_ * Tout;  // one block per (b, m-tile); 8 waves cover w

    // gated TCN, both streams
    k_gated<<<(nGated + 255) / 256, 256, 0, stream>>>(
        x0c, Tin, d, filt_w + (size_t)(l * 2 + 0) * DC * RC * 2,
        filt_b + (l * 2 + 0) * DC, gate_w + (size_t)(l * 2 + 0) * DC * RC * 2,
        gate_b + (l * 2 + 0) * DC, gx0, gx0T, Tout);
    k_gated<<<(nGated + 255) / 256, 256, 0, stream>>>(
        x1c, Tin, d, filt_w + (size_t)(l * 2 + 1) * DC * RC * 2,
        filt_b + (l * 2 + 1) * DC, gate_w + (size_t)(l * 2 + 1) * DC * RC * 2,
        gate_b + (l * 2 + 1) * DC, gx1, gx1T, Tout);

    // skip connection from stream 0
    k_skip<<<(B_ * SC * N_ * Tout + 255) / 256, 256, 0, stream>>>(
        gx0, Tout, skip_w + (size_t)l * SC * DC, skip_b + l * SC, skipB,
        skoff[l]);

    // g0 = gcn(gx0, A1)
    k_diff_wmma<<<nDiffB, 256, 0, stream>>>(A1, gx0T, h1T, Tout);
    k_diff_wmma<<<nDiffB, 256, 0, stream>>>(A1, h1T, h2T, Tout);
    k_gcn_pw<<<(nPW + 255) / 256, 256, 0, stream>>>(
        gx0T, h1T, h2T, Tout, gconv_w + (size_t)(l * 2 + 0) * RC * CIN,
        gconv_b + (l * 2 + 0) * RC, g0f, (__bf16*)nullptr);

    // g1 = gcn(gx1, A1) -> bf16 only
    k_diff_wmma<<<nDiffB, 256, 0, stream>>>(A1, gx1T, h1T, Tout);
    k_diff_wmma<<<nDiffB, 256, 0, stream>>>(A1, h1T, h2T, Tout);
    k_gcn_pw<<<(nPW + 255) / 256, 256, 0, stream>>>(
        gx1T, h1T, h2T, Tout, gconv_w + (size_t)(l * 2 + 1) * RC * CIN,
        gconv_b + (l * 2 + 1) * RC, (float*)nullptr, g1T);

    // xap = gcn(g1, A2)
    k_diff_wmma<<<nDiffB, 256, 0, stream>>>(A2, g1T, h1T, Tout);
    k_diff_wmma<<<nDiffB, 256, 0, stream>>>(A2, h1T, h2T, Tout);
    k_gcn_pw<<<(nPW + 255) / 256, 256, 0, stream>>>(
        g1T, h1T, h2T, Tout, ga_w + (size_t)l * RC * CIN, ga_b + l * RC, xapf,
        (__bf16*)nullptr);

    // residual + BN update (ping-pong)
    k_update<<<(B_ * RC * N_ * Tout + 255) / 256, 256, 0, stream>>>(
        g0f, xapf, x0c, x1c, Tin, Tout, x0n, x1n);
    { float* t0 = x0c; x0c = x0n; x0n = t0; }
    { float* t1 = x1c; x1c = x1n; x1n = t1; }
    Tin = Tout;
  }

  // end MLP
  k_end<<<(B_ * N_ + 255) / 256, 256, 0, stream>>>(skipB, end1_w, end1_b,
                                                   end2_w, end2_b, out);
}
